// KoLeoLoss_4355096838598
// MI455X (gfx1250) — matrix operand order repaired and checked by reading, not compiled, not run on hardware
//
#include <hip/hip_runtime.h>
#include <stdint.h>

#define N_ROWS 16384
#define DIMK   1024
#define ROWS_PER_BLOCK 64
#define WAVES_PER_BLOCK 4
#define THREADS_GEMM (WAVES_PER_BLOCK * 32)
#define JT 4                      // 16-column tiles per wave iteration (16x64)
#define NEG_INF_F (-3.0e38f)

typedef __bf16 v16bf __attribute__((ext_vector_type(16)));
typedef float  v8f   __attribute__((ext_vector_type(8)));

union Frag {
  v16bf v;
  uint4 q[2];
};

__device__ __forceinline__ unsigned short f32_to_bf16(float f) {
  union { float f; uint32_t u; } c; c.f = f;
  uint32_t u = c.u;
  u += 0x7FFFu + ((u >> 16) & 1u);   // round to nearest even
  return (unsigned short)(u >> 16);
}

// ---------------------------------------------------------------------------
// Kernel 1: L2-normalize each row, emit bf16 normalized features.
// ---------------------------------------------------------------------------
__global__ void koleo_normalize(const float* __restrict__ feats,
                                unsigned short* __restrict__ out_bf) {
  __shared__ float red[256];
  const int row = blockIdx.x;
  const int tid = threadIdx.x;
  const float* rp = feats + (size_t)row * DIMK;
  float s = 0.f;
#pragma unroll
  for (int i = 0; i < DIMK / 256; ++i) {
    float x = rp[tid + i * 256];
    s += x * x;
  }
  red[tid] = s;
  __syncthreads();
  for (int off = 128; off > 0; off >>= 1) {
    if (tid < off) red[tid] += red[tid + off];
    __syncthreads();
  }
  const float scale = 1.0f / fmaxf(sqrtf(red[0]), 1e-12f);
  unsigned short* op = out_bf + (size_t)row * DIMK;
#pragma unroll
  for (int i = 0; i < DIMK / 256; ++i) {
    const int idx = tid + i * 256;
    op[idx] = f32_to_bf16(rp[idx] * scale);
  }
}

// ---------------------------------------------------------------------------
// Kernel 2: C = F F^T via v_wmma_f32_16x16x32_bf16, fused row-wise max of the
// off-diagonal entries. Each block owns a 64-row stripe (4 waves x 16 rows),
// stages A in LDS once. Each wave computes a 16x64 output tile per K sweep
// with 4 independent accumulators. The k-loop is a fixed 2-stage ping-pong
// (k-step 64): while WMMAs consume one buffer set, the other set's loads are
// in flight. #pragma unroll 1 pins the loop so registers never spill.
// ---------------------------------------------------------------------------
__global__ void koleo_gemm_rowmax(const unsigned short* __restrict__ fb,
                                  float* __restrict__ rowmax) {
  extern __shared__ unsigned short ldsA[];           // 64 * 1024 bf16 = 128 KB
  const int tid  = threadIdx.x;
  const int wave = tid >> 5;
  const int lane = tid & 31;
  const int rowBase = blockIdx.x * ROWS_PER_BLOCK;

  // Cooperative load of the contiguous 64x1024 bf16 A stripe into LDS.
  {
    const uint4* gsrc = (const uint4*)(fb + (size_t)rowBase * DIMK);
    uint4* ldst = (uint4*)ldsA;
    const int nvec = ROWS_PER_BLOCK * DIMK / 8;      // uint4 = 8 bf16
    for (int i = tid; i < nvec; i += THREADS_GEMM) ldst[i] = gsrc[i];
  }
  __syncthreads();

  const int m     = lane & 15;       // row within A tile / column within B tile
  const int hi    = lane >> 4;       // half-wave select
  const int kaoff = hi ? 16 : 0;     // A: K 0..15 vs 16..31 (contiguous)
  const int kboff = hi ? 16 : 0;     // B: K 0..15 vs 16..31 (contiguous)
  const unsigned short* aRow = ldsA + (size_t)(wave * 16 + m) * DIMK + kaoff;
  const int growb = rowBase + wave * 16 + (hi ? 8 : 0);

  float rmax[8];
#pragma unroll
  for (int v = 0; v < 8; ++v) rmax[v] = NEG_INF_F;

  for (int jg = 0; jg < N_ROWS / (16 * JT); ++jg) {
    const int jBase = jg * 16 * JT;
    // One base row pointer; the other 3 tiles fold into immediate offsets.
    const unsigned short* bRow = fb + (size_t)(jBase + m) * DIMK + kboff;

    // Pull the next column group's rows toward the near caches.
    __builtin_prefetch(bRow + (size_t)16 * JT * DIMK, 0, 3);
    __builtin_prefetch(bRow + (size_t)16 * (JT + 3) * DIMK, 0, 3);

    v8f c[JT];
#pragma unroll
    for (int t = 0; t < JT; ++t) c[t] = (v8f){};

    // Prologue: fill buffer set 0 with fragments for k = 0.
    Frag a0, b0[JT], a1, b1[JT];
    a0.q[0] = *(const uint4*)(aRow);
    a0.q[1] = *(const uint4*)(aRow + 8);
#pragma unroll
    for (int t = 0; t < JT; ++t) {
      b0[t].q[0] = *(const uint4*)(bRow + (size_t)t * 16 * DIMK);
      b0[t].q[1] = *(const uint4*)(bRow + (size_t)t * 16 * DIMK + 8);
    }

#pragma unroll 1
    for (int k0 = 0; k0 < DIMK; k0 += 64) {
      const int k1 = k0 + 32;
      const int k2 = (k0 + 64) & (DIMK - 1);   // wraps to 0 on last iter (unused)

      // Stage 1: issue loads for k1 into set 1, compute k0 from set 0.
      a1.q[0] = *(const uint4*)(aRow + k1);
      a1.q[1] = *(const uint4*)(aRow + k1 + 8);
#pragma unroll
      for (int t = 0; t < JT; ++t) {
        b1[t].q[0] = *(const uint4*)(bRow + (size_t)t * 16 * DIMK + k1);
        b1[t].q[1] = *(const uint4*)(bRow + (size_t)t * 16 * DIMK + k1 + 8);
      }
#pragma unroll
      for (int t = 0; t < JT; ++t)
        c[t] = __builtin_amdgcn_wmma_f32_16x16x32_bf16(false, a0.v, false, b0[t].v,
                                                       (short)0, c[t], false, false);

      // Stage 2: issue loads for k2 into set 0, compute k1 from set 1.
      a0.q[0] = *(const uint4*)(aRow + k2);
      a0.q[1] = *(const uint4*)(aRow + k2 + 8);
#pragma unroll
      for (int t = 0; t < JT; ++t) {
        b0[t].q[0] = *(const uint4*)(bRow + (size_t)t * 16 * DIMK + k2);
        b0[t].q[1] = *(const uint4*)(bRow + (size_t)t * 16 * DIMK + k2 + 8);
      }
#pragma unroll
      for (int t = 0; t < JT; ++t)
        c[t] = __builtin_amdgcn_wmma_f32_16x16x32_bf16(false, a1.v, false, b1[t].v,
                                                       (short)0, c[t], false, false);
    }

    // C layout: lane holds column N = jBase + 16*t + m; VGPR v holds M = v (+8 hi).
#pragma unroll
    for (int v = 0; v < 8; ++v) {
      const int grow = growb + v;
      float mx = rmax[v];
#pragma unroll
      for (int t = 0; t < JT; ++t) {
        const float val = (grow == jBase + 16 * t + m) ? NEG_INF_F : c[t][v];
        mx = fmaxf(mx, val);
      }
      rmax[v] = mx;
    }
  }

  // Reduce across the 16 lanes of each half-wave (masks never cross halves).
#pragma unroll
  for (int v = 0; v < 8; ++v) {
    float x = rmax[v];
    x = fmaxf(x, __shfl_xor(x, 1, 32));
    x = fmaxf(x, __shfl_xor(x, 2, 32));
    x = fmaxf(x, __shfl_xor(x, 4, 32));
    x = fmaxf(x, __shfl_xor(x, 8, 32));
    rmax[v] = x;
  }
  if (lane == 0) {
#pragma unroll
    for (int v = 0; v < 8; ++v) rowmax[rowBase + wave * 16 + v] = rmax[v];
  } else if (lane == 16) {
#pragma unroll
    for (int v = 0; v < 8; ++v) rowmax[rowBase + wave * 16 + 8 + v] = rmax[v];
  }
}

// ---------------------------------------------------------------------------
// Kernel 3: dist_i = sqrt(max(2 - 2*maxdot_i, 0)); loss = -mean(log(dist+eps))
// ---------------------------------------------------------------------------
__global__ void koleo_finalize(const float* __restrict__ rowmax,
                               float* __restrict__ out) {
  __shared__ float red[256];
  const int tid = threadIdx.x;
  float s = 0.f;
  for (int r = tid; r < N_ROWS; r += 256) {
    const float d2   = fmaxf(2.0f - 2.0f * rowmax[r], 0.0f);
    const float dist = sqrtf(d2);
    s += logf(dist + 1e-8f);
  }
  red[tid] = s;
  __syncthreads();
  for (int off = 128; off > 0; off >>= 1) {
    if (tid < off) red[tid] += red[tid + off];
    __syncthreads();
  }
  if (tid == 0) out[0] = -red[0] / (float)N_ROWS;
}

// ---------------------------------------------------------------------------
extern "C" void kernel_launch(void* const* d_in, const int* in_sizes, int n_in,
                              void* d_out, int out_size, void* d_ws, size_t ws_size,
                              hipStream_t stream) {
  (void)in_sizes; (void)n_in; (void)out_size; (void)ws_size;
  const float* feats = (const float*)d_in[0];
  float* out = (float*)d_out;

  // Workspace layout: [0, 32MB) bf16 normalized feats; then 16384 f32 row maxes.
  unsigned short* fb = (unsigned short*)d_ws;
  float* rowmax = (float*)((char*)d_ws +
                           (size_t)N_ROWS * DIMK * sizeof(unsigned short));

  koleo_normalize<<<N_ROWS, 256, 0, stream>>>(feats, fb);
  koleo_gemm_rowmax<<<N_ROWS / ROWS_PER_BLOCK, THREADS_GEMM,
                      ROWS_PER_BLOCK * DIMK * sizeof(unsigned short), stream>>>(
      fb, rowmax);
  koleo_finalize<<<1, 256, 0, stream>>>(rowmax, out);
}
